// SNN_55267639164957
// MI455X (gfx1250) — compile-verified, matrix operand order checked by reading, not run on hardware
//
#include <hip/hip_runtime.h>
#include <hip/hip_bf16.h>

// ---------------------------------------------------------------------------
// Fused CobaLIF SNN for gfx1250 (MI455X), wave32 + WMMA bf16.
//   kernel 1: pack/split weights -> bf16 B-operand tiles in d_ws (~1MB, L2-resident)
//   kernel 2: persistent fused scan, 16 WGs (one per 16-batch tile), 8 waves each.
// Round-3 changes: persistent neuron state (g_e, g_i, v) moved to LDS in a
// per-thread-contiguous [nt][lane][8] layout (ds_load_b128-able) so it no
// longer lives in VGPRs across the WMMA loops -> kills the scratch spills
// that sat on the per-step critical path.
// ---------------------------------------------------------------------------

typedef __bf16 bf16_t;
typedef __attribute__((ext_vector_type(16))) __bf16 v16bf;
typedef __attribute__((ext_vector_type(8)))  __bf16 v8bf;
typedef __attribute__((ext_vector_type(8)))  float   v8f;

#define WMMA_BF16(A, Bm, C) \
  __builtin_amdgcn_wmma_f32_16x16x32_bf16(false, (A), false, (Bm), (short)0, (C), false, false)

// problem sizes
constexpr int B_ = 256, T_ = 1000, F_ = 700, H_ = 256, O_ = 20;
constexpr int FP    = 704;        // F padded to multiple of 32
constexpr int KT_IN = FP / 32;    // 22 k-tiles for feed-forward
constexpr int KT_H  = H_ / 32;    // 8  k-tiles over H
constexpr int NT_H  = H_ / 16;    // 16 n-tiles over H
constexpr int NT_O  = 2;          // O padded to 32

constexpr int    TILE   = 512;    // packed B tile: 32 lanes x 16 bf16
constexpr size_t SZ_IN  = (size_t)KT_IN * NT_H * TILE;
constexpr size_t SZ_REC = (size_t)KT_H  * NT_H * TILE;
constexpr size_t SZ_OUT = (size_t)KT_H  * NT_O * TILE;
constexpr size_t TOT_PACK = 2 * SZ_IN + 2 * SZ_REC + SZ_OUT;

// ---------------------------------------------------------------------------
// B-operand packing: for a 32x16 (KxN) bf16 tile, lane L holds n = L&15 and
// 16 K-values in two 8-element runs (mirrors the documented 16-bit A layout):
//   elems 0..7  -> k = half*8 + e
//   elems 8..15 -> k = 16 + half*8 + (e-8)        (half = L>>4)
// Stored per-lane contiguous (32B) so the hot loop loads 2x b128 per fragment.
// ---------------------------------------------------------------------------
__device__ __forceinline__ void decode_pack(int r, int NT, int& kt, int& n, int& kk) {
  int e    = r & 15;
  int lane = (r >> 4) & 31;
  int tile = r >> 9;
  int nt   = tile % NT;
  kt       = tile / NT;
  n        = nt * 16 + (lane & 15);
  int hf   = lane >> 4;
  kk = (e < 8) ? (hf * 8 + e) : (16 + hf * 8 + (e - 8));
}

__global__ void snn_pack_weights(const float* __restrict__ w_in,
                                 const float* __restrict__ w_rec,
                                 const float* __restrict__ w_out,
                                 bf16_t* __restrict__ pk) {
  size_t idx = (size_t)blockIdx.x * blockDim.x + threadIdx.x;
  if (idx >= TOT_PACK) return;
  float val = 0.0f;
  if (idx < 2 * SZ_IN) {                       // w_in_e / w_in_i  (ReLU split)
    bool inh = idx >= SZ_IN;
    int r = (int)(inh ? idx - SZ_IN : idx);
    int kt, n, kk; decode_pack(r, NT_H, kt, n, kk);
    int k = kt * 32 + kk;
    if (k < F_) {
      float w = w_in[(size_t)n * F_ + k];
      val = inh ? fmaxf(-w, 0.0f) : fmaxf(w, 0.0f);
    }
  } else if (idx < 2 * SZ_IN + 2 * SZ_REC) {   // w_rec_e / w_rec_i
    size_t b = idx - 2 * SZ_IN;
    bool inh = b >= SZ_REC;
    int r = (int)(inh ? b - SZ_REC : b);
    int kt, n, kk; decode_pack(r, NT_H, kt, n, kk);
    int k = kt * 32 + kk;
    float w = w_rec[(size_t)n * H_ + k];
    val = inh ? fmaxf(-w, 0.0f) : fmaxf(w, 0.0f);
  } else {                                     // w_out (signed, N padded to 32)
    int r = (int)(idx - 2 * SZ_IN - 2 * SZ_REC);
    int kt, n, kk; decode_pack(r, NT_O, kt, n, kk);
    int k = kt * 32 + kk;
    if (n < O_) val = w_out[(size_t)n * H_ + k];
  }
  pk[idx] = (bf16_t)val;
}

// 16B+16B fragment gather (pointer already lane-adjusted, 16B apart halves).
__device__ __forceinline__ v16bf load_frag(const bf16_t* p) {
  union { v16bf v; v8bf h[2]; } u;
  u.h[0] = *(const v8bf*)(p);
  u.h[1] = *(const v8bf*)(p + 8);
  return u.v;
}

// A fragment (16x32 bf16, M x K) from a row-major LDS matrix [16][stride].
__device__ __forceinline__ v16bf load_a(const bf16_t* base, int stride, int k0, int lane) {
  const int m  = lane & 15;
  const int hf = lane >> 4;
  const bf16_t* p = base + m * stride + k0 + hf * 8;
  union { v16bf v; v8bf h[2]; } u;
  u.h[0] = *(const v8bf*)(p);
  u.h[1] = *(const v8bf*)(p + 16);
  return u.v;
}

__global__ __launch_bounds__(256, 1)
void snn_fused(const float* __restrict__ x,       // [B,T,F] f32 (binary spikes)
               const bf16_t* __restrict__ pk,     // packed weights
               float* __restrict__ volt,          // [T,B,O] f32
               float* __restrict__ spk) {         // [T,B,H] f32
  __shared__ __align__(16) bf16_t lds_x[2 * 16 * FP];  // 45 KB  x slices (double buffer)
  __shared__ __align__(16) bf16_t lds_z[2 * 16 * H_];  // 16 KB  spikes (double buffer)
  // persistent neuron state, per-thread-contiguous: [nt(16)][lane(32)][r(8)] f32
  __shared__ __align__(16) float lds_ge[16 * 32 * 8];  // 16 KB
  __shared__ __align__(16) float lds_gi[16 * 32 * 8];  // 16 KB
  __shared__ __align__(16) float lds_vm[16 * 32 * 8];  // 16 KB
  __shared__ float lds_vo[16 * 32];                    //  2 KB  LI membrane
  __shared__ float lds_io[16 * 32];                    //  2 KB  LI synapse

  const int tid  = threadIdx.x;
  const int wave = tid >> 5;
  const int lane = tid & 31;
  const int hf   = lane >> 4;
  const int nlo  = lane & 15;
  const int b0   = blockIdx.x * 16;

  const bf16_t* pk_in_e  = pk;
  const bf16_t* pk_in_i  = pk_in_e  + SZ_IN;
  const bf16_t* pk_rec_e = pk_in_i  + SZ_IN;
  const bf16_t* pk_rec_i = pk_rec_e + SZ_REC;
  const bf16_t* pk_out   = pk_rec_i + SZ_REC;

  // prologue: zero state + z buffer 0 + LI state, stage x[:, t=0, :] into buffer 0
  for (int i = tid; i < 16 * H_; i += 256) lds_z[i] = (bf16_t)0.0f;
  for (int i = tid; i < 16 * 32 * 8; i += 256) { lds_ge[i] = 0.f; lds_gi[i] = 0.f; lds_vm[i] = 0.f; }
  for (int i = tid; i < 16 * 32; i += 256) { lds_vo[i] = 0.0f; lds_io[i] = 0.0f; }
#pragma unroll 1
  for (int m = 0; m < 16; ++m) {
    const float* row = x + ((size_t)(b0 + m) * T_ + 0) * F_;
    bf16_t* dst = lds_x + m * FP;
    for (int f = tid; f < FP; f += 256) dst[f] = (bf16_t)((f < F_) ? row[f] : 0.0f);
  }
  __syncthreads();

  for (int t = 0; t < T_; ++t) {
    const int cur = t & 1, nxt = cur ^ 1;
    const bf16_t* xb   = lds_x + cur * (16 * FP);
    const bf16_t* zb   = lds_z + cur * (16 * H_);
    bf16_t*       zb_n = lds_z + nxt * (16 * H_);

    // ---- per n-tile: accumulate drive, update neuron, retire immediately ----
#pragma unroll 1
    for (int nt2 = 0; nt2 < 2; ++nt2) {
      const int nt = wave * 2 + nt2;
      v8f ae = {0, 0, 0, 0, 0, 0, 0, 0};
      v8f ai = {0, 0, 0, 0, 0, 0, 0, 0};
      const bf16_t* be_p = pk_in_e + (size_t)nt * TILE + lane * 16;
      const bf16_t* bi_p = pk_in_i + (size_t)nt * TILE + lane * 16;
#pragma unroll 2
      for (int kt = 0; kt < KT_IN; ++kt) {                 // K = 704 feed-forward
        v16bf a  = load_a(xb, FP, kt * 32, lane);
        ae = WMMA_BF16(a, load_frag(be_p + (size_t)kt * NT_H * TILE), ae);
        ai = WMMA_BF16(a, load_frag(bi_p + (size_t)kt * NT_H * TILE), ai);
      }
      const bf16_t* re_p = pk_rec_e + (size_t)nt * TILE + lane * 16;
      const bf16_t* ri_p = pk_rec_i + (size_t)nt * TILE + lane * 16;
#pragma unroll 2
      for (int kt = 0; kt < KT_H; ++kt) {                  // K = 256 recurrent
        v16bf a  = load_a(zb, H_, kt * 32, lane);
        ae = WMMA_BF16(a, load_frag(re_p + (size_t)kt * NT_H * TILE), ae);
        ai = WMMA_BF16(a, load_frag(ri_p + (size_t)kt * NT_H * TILE), ai);
      }
      // neuron update (state streamed from LDS, b128-mergeable) + publish z_t
      const int n     = wave * 32 + nt2 * 16 + nlo;
      const int sbase = (nt * 32 + lane) * 8;
      float* sge = &lds_ge[sbase];
      float* sgi = &lds_gi[sbase];
      float* svm = &lds_vm[sbase];
#pragma unroll
      for (int r = 0; r < 8; ++r) {
        float e = sge[r] * 0.98f + ae[r];                      // 1 - DT/tau_e
        float i = sgi[r] * 0.99f + ai[r];                      // 1 - DT/tau_i
        float v = svm[r];
        v += 0.5f * (0.25f * (0.0f - v) + e * (60.0f - v) + i * (0.0f - v)); // DT/C_m
        float z = (v > 1.0f) ? 1.0f : 0.0f;                    // heaviside(v - 1)
        sge[r] = e; sgi[r] = i;
        svm[r] = (1.0f - z) * v;                               // reset to 0
        const int m = r + 8 * hf;
        zb_n[m * H_ + n] = (bf16_t)z;
        spk[((size_t)t * B_ + (b0 + m)) * H_ + n] = z;
      }
    }

    // ---- stage x_{t+1} into the other buffer (overlaps with compute waves) ----
    if (t + 1 < T_) {
      bf16_t* xdst = lds_x + nxt * (16 * FP);
#pragma unroll 1
      for (int m = 0; m < 16; ++m) {
        const float* row = x + ((size_t)(b0 + m) * T_ + (t + 1)) * F_;
        bf16_t* dst = xdst + m * FP;
        for (int f = tid; f < FP; f += 256) dst[f] = (bf16_t)((f < F_) ? row[f] : 0.0f);
      }
    }

    __syncthreads();   // z_t and x_{t+1} buffers complete

    // ---- readout (wave 0): i_in = z_t @ w_out^T, LI update, store voltages ----
    // Safe without a second barrier: next iteration writes only the *other*
    // z/x buffers, while this reads zb_n (= next iteration's read buffer).
    if (wave == 0) {
      v8f a0 = {0, 0, 0, 0, 0, 0, 0, 0};
      v8f a1 = {0, 0, 0, 0, 0, 0, 0, 0};
      const bf16_t* bo_p = pk_out + lane * 16;
#pragma unroll 2
      for (int kt = 0; kt < KT_H; ++kt) {
        v16bf a = load_a(zb_n, H_, kt * 32, lane);
        a0 = WMMA_BF16(a, load_frag(bo_p + (size_t)(kt * NT_O + 0) * TILE), a0);
        a1 = WMMA_BF16(a, load_frag(bo_p + (size_t)(kt * NT_O + 1) * TILE), a1);
      }
#pragma unroll
      for (int nt2 = 0; nt2 < 2; ++nt2) {
        const int o = nt2 * 16 + nlo;
#pragma unroll
        for (int r = 0; r < 8; ++r) {
          const int m   = r + 8 * hf;
          const int li  = m * 32 + o;
          float iin = (nt2 == 0) ? a0[r] : a1[r];
          float vvo = lds_vo[li], vio = lds_io[li];
          float vv  = vvo + 0.01f * (vio - vvo);   // vo uses old io (DT/tau_mem)
          float ii  = vio * 0.98f + iin;           // io decays (1 - DT/tau_syn) + input
          lds_vo[li] = vv; lds_io[li] = ii;
          if (o < O_) volt[((size_t)t * B_ + (b0 + m)) * O_ + o] = vv;
        }
      }
    }
  }
}

extern "C" void kernel_launch(void* const* d_in, const int* in_sizes, int n_in,
                              void* d_out, int out_size, void* d_ws, size_t ws_size,
                              hipStream_t stream) {
  (void)in_sizes; (void)n_in; (void)out_size; (void)ws_size;
  const float* x     = (const float*)d_in[0];
  const float* w_in  = (const float*)d_in[1];
  const float* w_rec = (const float*)d_in[2];
  const float* w_out = (const float*)d_in[3];

  float* volt = (float*)d_out;                       // [T,B,O]
  float* spk  = volt + (size_t)T_ * B_ * O_;         // [T,B,H]
  bf16_t* pkw = (bf16_t*)d_ws;                       // ~1 MB packed weights

  const int nthr = 256;
  const int nblk = (int)((TOT_PACK + nthr - 1) / nthr);
  snn_pack_weights<<<nblk, nthr, 0, stream>>>(w_in, w_rec, w_out, pkw);
  snn_fused<<<B_ / 16, 256, 0, stream>>>(x, pkw, volt, spk);
}
